// GenesisV1_56109452755245
// MI455X (gfx1250) — matrix operand order repaired and checked by reading, not compiled, not run on hardware
//
#include <hip/hip_runtime.h>
#include <cstdint>

// ---------------------------------------------------------------------------
// Shapes
// ---------------------------------------------------------------------------
#define VOCAB    32000
#define DMODEL   512
#define NNODES   64
#define BATCH    8
#define SEQ      256
#define TOKENS   (BATCH * SEQ)      // 2048
#define KTILES   (DMODEL / 32)      // 16
#define NB_STRIDE 520               // bf16 node buffer row stride (pad vs 512)

typedef __attribute__((ext_vector_type(16))) __bf16 v16bf;
typedef __attribute__((ext_vector_type(8)))  __bf16 v8bf;
typedef __attribute__((ext_vector_type(8)))  float  v8f;
typedef uint32_t u32x4 __attribute__((ext_vector_type(4)));
typedef uint32_t u32x8 __attribute__((ext_vector_type(8)));

__device__ __forceinline__ v8f wmma_bf16(v16bf a, v16bf b, v8f c) {
    // v_wmma_f32_16x16x32_bf16  (neg_a, A, neg_b, B, c_mod, C, reuse_a, reuse_b)
    return __builtin_amdgcn_wmma_f32_16x16x32_bf16(false, a, false, b,
                                                   (short)0, c, false, false);
}

__device__ __forceinline__ v16bf cat8(v8bf lo, v8bf hi) {
    return __builtin_shufflevector(lo, hi, 0,1,2,3,4,5,6,7,8,9,10,11,12,13,14,15);
}

__device__ __forceinline__ v8f zero8() {
    v8f z;
#pragma unroll
    for (int i = 0; i < 8; ++i) z[i] = 0.0f;
    return z;
}

// A fragment (16x32 bf16, row-major source with given element stride).
// Lane L: row = L%16 ; K chunks: [kt*32 + 8*(L/16) .. +7] and [+16 .. +23].
__device__ __forceinline__ v16bf load_a_frag(const __bf16* base, int stride,
                                             int mt, int kt, int lane) {
    int r  = mt * 16 + (lane & 15);
    int k0 = kt * 32 + ((lane >> 4) << 3);
    const __bf16* p = base + r * stride + k0;
    v8bf lo = *(const v8bf*)(p);
    v8bf hi = *(const v8bf*)(p + 16);
    return cat8(lo, hi);
}

// B fragment from pre-swizzled weights: per (nt,kt) tile, lane L owns 16
// contiguous bf16 (32 bytes) -> two global_load_b128 per lane.
__device__ __forceinline__ v16bf load_b_frag(const __bf16* wswz, long nt,
                                             int kt, int lane) {
    const __bf16* p = wswz + (((nt * 16 + kt) * 32L) + lane) * 16;
    return *(const v16bf*)p;
}

__device__ __forceinline__ float sigmoidf_fast(float x) {
    return 1.0f / (1.0f + __expf(-x));
}
__device__ __forceinline__ float tanhf_fast(float x) {
    return 1.0f - 2.0f / (__expf(2.0f * x) + 1.0f);
}

// ---------------------------------------------------------------------------
// TDM: 2D tensor load (global -> LDS) via Tensor Data Mover.
// D# per CDNA5 ISA: group0 = {count|flags, lds_addr,
// global_addr[31:0], global_addr[56:32]|type<<30}; group1 packs data_size,
// tensor_dim0/1, tile_dim0/1, dim0 stride. Tile == whole tensor here.
// ---------------------------------------------------------------------------
__device__ __forceinline__ void tdm_load_2d(const void* gptr, uint32_t lds_off,
                                            uint32_t dim0, uint32_t dim1,
                                            uint32_t stride0, uint32_t dsz_log) {
    uint64_t ga = (uint64_t)gptr;
    u32x4 g0;
    g0[0] = 1u;                                   // count=1, user descriptor
    g0[1] = lds_off;                              // LDS byte address
    g0[2] = (uint32_t)ga;                         // global addr [31:0]
    g0[3] = (uint32_t)(ga >> 32) | (2u << 30);    // addr[56:32] | type=2
    u32x8 g1;
    g1[0] = dsz_log << 16;                        // wg_mask=0, data_size
    g1[1] = (dim0 & 0xFFFFu) << 16;               // tensor_dim0 [79:48]
    g1[2] = (dim0 >> 16) | ((dim1 & 0xFFFFu) << 16);   // tensor_dim1 [111:80]
    g1[3] = (dim1 >> 16) | ((dim0 & 0xFFFFu) << 16);   // tile_dim0 [127:112]
    g1[4] = dim1 & 0xFFFFu;                       // tile_dim1 [143:128]
    g1[5] = stride0;                              // tensor_dim0_stride lo
    g1[6] = 0u;
    g1[7] = 0u;
    asm volatile("tensor_load_to_lds %0, %1" :: "s"(g0), "s"(g1) : "memory");
}

// ---------------------------------------------------------------------------
// Kernel 1: swizzle fp32 weight matrix [512 x N] into bf16 WMMA-B fragment
// layout: flat = ((nt*16 + kt)*32 + lane)*16 + i ; k = kt*32 + (lane/16)*8 + (i%8) + (i/8)*16
// ---------------------------------------------------------------------------
__global__ void k_swizzle_b(const float* __restrict__ src, int row_off, long N,
                            __bf16* __restrict__ dst) {
    long f = (long)blockIdx.x * blockDim.x + threadIdx.x;
    if (f >= 512L * N) return;
    int  i  = (int)(f & 15);
    int  L  = (int)((f >> 4) & 31);
    int  kt = (int)((f >> 9) & 15);
    long nt = f >> 13;
    int  k  = kt * 32 + ((L >> 4) << 3) + (i & 7) + ((i >> 3) << 4);
    long n  = nt * 16 + (L & 15);
    dst[f] = (__bf16)src[(long)(row_off + k) * N + n];
}

// ---------------------------------------------------------------------------
// Kernel 2: Xg/Xu = gather(embedding, idx) @ W_x + bias   (M=2048,K=512,N=512)
// One wave per 16x64 output strip (4 N-tiles share one gathered A fragment).
// ---------------------------------------------------------------------------
__global__ void k_xproj(const int* __restrict__ idx,
                        const float* __restrict__ emb,
                        const __bf16* __restrict__ wswz,
                        const float* __restrict__ bias,
                        float* __restrict__ out) {
    int lane = threadIdx.x;
    int mt = blockIdx.y;
    long ntBase = (long)blockIdx.x * 4;
    int h = lane >> 4, ln = lane & 15;
    int r = mt * 16 + ln;                 // token row
    const float* arow = emb + (long)idx[r] * DMODEL;
    v8f acc[4];
#pragma unroll
    for (int q = 0; q < 4; ++q) acc[q] = zero8();
    for (int kt = 0; kt < KTILES; ++kt) {
        int k0 = kt * 32 + h * 8;
        v16bf a;
#pragma unroll
        for (int i = 0; i < 8; ++i) {
            a[i]     = (__bf16)arow[k0 + i];
            a[i + 8] = (__bf16)arow[k0 + 16 + i];
        }
#pragma unroll
        for (int q = 0; q < 4; ++q) {
            v16bf b = load_b_frag(wswz, ntBase + q, kt, lane);
            acc[q] = wmma_bf16(a, b, acc[q]);
        }
    }
#pragma unroll
    for (int q = 0; q < 4; ++q) {
        int n = (int)(ntBase + q) * 16 + ln;
        float bn = bias[n];
#pragma unroll
        for (int j = 0; j < 8; ++j) {
            int m = mt * 16 + j + 8 * h;
            out[(long)m * DMODEL + n] = acc[q][j] + bn;
        }
    }
}

// ---------------------------------------------------------------------------
// Kernel 3: persistent recurrent scan. 1 workgroup == 1 batch, 16 waves.
// LDS: fp32 master state [64x512] + bf16 WMMA copy [64x520].
// Wave w owns N-strips nt = 2w, 2w+1 across all 4 M-tiles; gate and candidate
// GEMMs fused into ONE K-sweep (shared A fragments, dual accumulators).
// ---------------------------------------------------------------------------
__global__ __launch_bounds__(512, 1)
void k_recurrent(const float* __restrict__ nodes0,
                 const __bf16* __restrict__ WgN,
                 const __bf16* __restrict__ WuN,
                 const float* __restrict__ Xg,
                 const float* __restrict__ Xu,
                 float* __restrict__ cons) {
    extern __shared__ char smem[];
    float*  nf = (float*)smem;                               // 64*512 f32
    __bf16* nb = (__bf16*)(smem + NNODES * DMODEL * 4);      // 64*520 bf16

    const int tid  = threadIdx.x;
    const int b    = blockIdx.x;
    const int lane = tid & 31;
    const int wave = tid >> 5;
    const int h  = lane >> 4;
    const int ln = lane & 15;
    const int nt0 = wave * 2;

    // init: TDM bulk-copies nodes0 (64x512 f32, 128 KB) straight into LDS,
    // then all threads build the padded bf16 copy.
    if (wave == 0) {
        tdm_load_2d(nodes0, 0u, DMODEL, NNODES, DMODEL, 2u /* 4-byte */);
        __builtin_amdgcn_s_wait_tensorcnt(0);
    }
    __syncthreads();
    for (int e = tid; e < NNODES * DMODEL; e += 512)
        nb[(e >> 9) * NB_STRIDE + (e & 511)] = (__bf16)nf[e];
    __syncthreads();

    for (int t = 0; t < SEQ; ++t) {
        const float* xg = Xg + (long)(b * SEQ + t) * DMODEL;
        const float* xu = Xu + (long)(b * SEQ + t) * DMODEL;

        // ---- fused gate+candidate GEMMs: one K-sweep, shared A frags ----
        v8f accg[2][4], accu[2][4];
#pragma unroll
        for (int s = 0; s < 2; ++s)
#pragma unroll
            for (int mt = 0; mt < 4; ++mt) {
                accg[s][mt] = zero8();
                accu[s][mt] = zero8();
            }
        for (int kt = 0; kt < KTILES; ++kt) {
            v16bf a[4];
#pragma unroll
            for (int mt = 0; mt < 4; ++mt)
                a[mt] = load_a_frag(nb, NB_STRIDE, mt, kt, lane);
#pragma unroll
            for (int s = 0; s < 2; ++s) {
                v16bf bg_ = load_b_frag(WgN, nt0 + s, kt, lane);
#pragma unroll
                for (int mt = 0; mt < 4; ++mt)
                    accg[s][mt] = wmma_bf16(a[mt], bg_, accg[s][mt]);
                v16bf bu_ = load_b_frag(WuN, nt0 + s, kt, lane);
#pragma unroll
                for (int mt = 0; mt < 4; ++mt)
                    accu[s][mt] = wmma_bf16(a[mt], bu_, accu[s][mt]);
            }
        }

        // ---- nonlinearity + fp32 state update (exclusive ownership) ----
#pragma unroll
        for (int s = 0; s < 2; ++s) {
            int n = (nt0 + s) * 16 + ln;
            float xgv = xg[n];
            float xuv = xu[n];
#pragma unroll
            for (int mt = 0; mt < 4; ++mt)
#pragma unroll
                for (int j = 0; j < 8; ++j) {
                    int m = mt * 16 + j + 8 * h;
                    float g = sigmoidf_fast(accg[s][mt][j] + xgv);
                    float u = tanhf_fast(accu[s][mt][j] + xuv);
                    float o = nf[m * DMODEL + n];
                    nf[m * DMODEL + n] = g * u + (1.0f - g) * o;
                }
        }
        __syncthreads();   // all bf16 reads done, all fp32 updates done

        // refresh bf16 copy + per-column consensus mean
        for (int e = tid; e < NNODES * DMODEL; e += 512)
            nb[(e >> 9) * NB_STRIDE + (e & 511)] = (__bf16)nf[e];
        float s = 0.0f;
#pragma unroll 8
        for (int m = 0; m < NNODES; ++m) s += nf[m * DMODEL + tid];
        cons[(long)(b * SEQ + t) * DMODEL + tid] = s * (1.0f / NNODES);
        __syncthreads();
    }
}

// ---------------------------------------------------------------------------
// Kernel 4: row LayerNorm, fp32 -> bf16 (wave per row, shuffle reduction)
// ---------------------------------------------------------------------------
__global__ void k_layernorm(const float* __restrict__ cons,
                            const float* __restrict__ w,
                            const float* __restrict__ bvec,
                            __bf16* __restrict__ out) {
    int row  = blockIdx.x * 8 + (threadIdx.x >> 5);
    int lane = threadIdx.x & 31;
    const float* x = cons + (long)row * DMODEL;
    float v[16], s = 0.0f, s2 = 0.0f;
#pragma unroll
    for (int i = 0; i < 16; ++i) {
        v[i] = x[lane * 16 + i];
        s += v[i];
        s2 += v[i] * v[i];
    }
#pragma unroll
    for (int off = 16; off >= 1; off >>= 1) {
        s  += __shfl_xor(s,  off, 32);
        s2 += __shfl_xor(s2, off, 32);
    }
    float mu  = s  * (1.0f / DMODEL);
    float var = s2 * (1.0f / DMODEL) - mu * mu;
    float inv = rsqrtf(var + 1e-5f);
#pragma unroll
    for (int i = 0; i < 16; ++i) {
        int c = lane * 16 + i;
        out[(long)row * DMODEL + c] = (__bf16)((v[i] - mu) * inv * w[c] + bvec[c]);
    }
}

// ---------------------------------------------------------------------------
// Kernel 5: head GEMM  [2048,512]bf16 @ [512,32000]bf16 + bh -> fp32 logits
// 8 waves / block; each wave does a 16x80 strip (5 N-tiles share one A frag).
// grid (50, 128): 50*8*5*16 = 32000 columns, 128*16 = 2048 rows.
// ---------------------------------------------------------------------------
__global__ __launch_bounds__(256)
void k_head(const __bf16* __restrict__ A, const __bf16* __restrict__ Wswz,
            const float* __restrict__ bh, float* __restrict__ out) {
    int lane = threadIdx.x & 31;
    int w    = threadIdx.x >> 5;
    long ntBase = ((long)blockIdx.x * 8 + w) * 5;
    int  mt  = blockIdx.y;
    int h = lane >> 4, ln = lane & 15;
    const __bf16* arow = A + (long)(mt * 16 + ln) * DMODEL;
    v8f acc[5];
#pragma unroll
    for (int q = 0; q < 5; ++q) acc[q] = zero8();
    for (int kt = 0; kt < KTILES; ++kt) {
        int k0 = kt * 32 + h * 8;
        v8bf lo = *(const v8bf*)(arow + k0);
        v8bf hi = *(const v8bf*)(arow + k0 + 16);
        v16bf a = cat8(lo, hi);
#pragma unroll
        for (int q = 0; q < 5; ++q) {
            v16bf b = load_b_frag(Wswz, ntBase + q, kt, lane);
            acc[q] = wmma_bf16(a, b, acc[q]);
        }
    }
#pragma unroll
    for (int q = 0; q < 5; ++q) {
        long n = (ntBase + q) * 16 + ln;
        float bn = bh[n];
#pragma unroll
        for (int j = 0; j < 8; ++j) {
            long m = mt * 16 + j + 8 * h;
            out[m * VOCAB + n] = acc[q][j] + bn;
        }
    }
}

// ---------------------------------------------------------------------------
// Host orchestration
// ---------------------------------------------------------------------------
extern "C" void kernel_launch(void* const* d_in, const int* in_sizes, int n_in,
                              void* d_out, int out_size, void* d_ws, size_t ws_size,
                              hipStream_t stream) {
    (void)in_sizes; (void)n_in; (void)out_size; (void)ws_size;
    const int*   idx = (const int*)  d_in[0];
    const float* emb = (const float*)d_in[1];
    const float* mn  = (const float*)d_in[2];
    const float* Wg  = (const float*)d_in[3];
    const float* bg  = (const float*)d_in[4];
    const float* Wu  = (const float*)d_in[5];
    const float* bu  = (const float*)d_in[6];
    const float* lnw = (const float*)d_in[7];
    const float* lnb = (const float*)d_in[8];
    const float* Wh  = (const float*)d_in[9];
    const float* bh  = (const float*)d_in[10];
    float* out = (float*)d_out;

    // workspace carve-out
    char* ws = (char*)d_ws;
    size_t off = 0;
    auto carve = [&](size_t bytes) -> char* {
        char* p = ws + off;
        off = (off + bytes + 255) & ~(size_t)255;
        return p;
    };
    __bf16* wgx_s = (__bf16*)carve((size_t)512 * 512 * 2);
    __bf16* wgn_s = (__bf16*)carve((size_t)512 * 512 * 2);
    __bf16* wux_s = (__bf16*)carve((size_t)512 * 512 * 2);
    __bf16* wun_s = (__bf16*)carve((size_t)512 * 512 * 2);
    __bf16* wh_s  = (__bf16*)carve((size_t)512 * VOCAB * 2);
    float*  Xg    = (float*) carve((size_t)TOKENS * DMODEL * 4);
    float*  Xu    = (float*) carve((size_t)TOKENS * DMODEL * 4);
    float*  cons  = (float*) carve((size_t)TOKENS * DMODEL * 4);
    __bf16* lnout = (__bf16*)carve((size_t)TOKENS * DMODEL * 2);

    // 1) swizzle weights into WMMA B-fragment layout (bf16)
    {
        long tot = 512L * 512;
        int  g   = (int)((tot + 255) / 256);
        k_swizzle_b<<<g, 256, 0, stream>>>(Wg,   0, 512, wgx_s);
        k_swizzle_b<<<g, 256, 0, stream>>>(Wg, 512, 512, wgn_s);
        k_swizzle_b<<<g, 256, 0, stream>>>(Wu,   0, 512, wux_s);
        k_swizzle_b<<<g, 256, 0, stream>>>(Wu, 512, 512, wun_s);
        long toth = 512L * VOCAB;
        int  gh   = (int)((toth + 255) / 256);
        k_swizzle_b<<<gh, 256, 0, stream>>>(Wh, 0, VOCAB, wh_s);
    }

    // 2) time-parallel input projections (embedding gather fused)
    {
        dim3 grid(DMODEL / 16 / 4, TOKENS / 16);
        k_xproj<<<grid, 32, 0, stream>>>(idx, emb, wgx_s, bg, Xg);
        k_xproj<<<grid, 32, 0, stream>>>(idx, emb, wux_s, bu, Xu);
    }

    // 3) persistent recurrent scan: 1 WG / batch, 198 KB LDS
    {
        const int smem = NNODES * DMODEL * 4 + NNODES * NB_STRIDE * 2; // 197632
        hipFuncSetAttribute(reinterpret_cast<const void*>(k_recurrent),
                            hipFuncAttributeMaxDynamicSharedMemorySize, smem);
        k_recurrent<<<BATCH, 512, smem, stream>>>(mn, wgn_s, wun_s, Xg, Xu, cons);
    }

    // 4) layernorm -> bf16
    k_layernorm<<<TOKENS / 8, 256, 0, stream>>>(cons, lnw, lnb, lnout);

    // 5) head GEMM -> fp32 logits
    {
        dim3 grid(VOCAB / 16 / 8 / 5, TOKENS / 16);
        k_head<<<grid, 256, 0, stream>>>(lnout, wh_s, bh, out);
    }
}
